// RationalLinearSplineFlow_77927886618676
// MI455X (gfx1250) — compile-verified, matrix operand order checked by reading, not run on hardware
//
#include <hip/hip_runtime.h>
#include <hip/hip_bf16.h>

typedef __attribute__((ext_vector_type(16))) __bf16 v16bf;
typedef __attribute__((ext_vector_type(8)))  float  v8f;

#define DM 128         // D_MODEL
#define NP 64          // padded param channels (63 real + 1 zero)
#define TB 20.0f       // TAIL_BOUND
#define MBW 1e-3f      // MIN_BIN_WIDTH
#define MBH 1e-3f      // MIN_BIN_HEIGHT
#define MDER 1e-3f     // MIN_DERIVATIVE
#define SEPS 1e-6f     // EPS
#define LOG_STRIDE 65  // 64 + 1 pad -> conflict-free per-token LDS reads

__device__ __forceinline__ float splus(float x) {
    // numerically stable softplus: max(x,0) + log1p(exp(-|x|))
    return fmaxf(x, 0.f) + log1pf(expf(-fabsf(x)));
}
__device__ __forceinline__ float sigm(float x) {
    return 1.f / (1.f + expf(-x));
}
// packed weight row n (0..63), feature k: [Ww(16) | Wh(16) | Wd(15) | Wl(16) | 0]
__device__ __forceinline__ float wrow(int n, int k, const float* Ww, const float* Wh,
                                      const float* Wd, const float* Wl) {
    if (n < 16) return Ww[n * DM + k];
    if (n < 32) return Wh[(n - 16) * DM + k];
    if (n < 47) return Wd[(n - 32) * DM + k];
    if (n < 63) return Wl[(n - 47) * DM + k];
    return 0.f;
}
__device__ __forceinline__ float brow(int n, const float* bw, const float* bh,
                                      const float* bd, const float* bl) {
    if (n < 16) return bw[n];
    if (n < 32) return bh[n - 16];
    if (n < 47) return bd[n - 32];
    if (n < 63) return bl[n - 47];
    return 0.f;
}

__global__ __launch_bounds__(256) void rls_flow_kernel(
    const float* __restrict__ inp, const float* __restrict__ cond,
    const float* __restrict__ Ww, const float* __restrict__ bw,
    const float* __restrict__ Wh, const float* __restrict__ bh,
    const float* __restrict__ Wd, const float* __restrict__ bd,
    const float* __restrict__ Wl, const float* __restrict__ bl,
    float* __restrict__ outp, float* __restrict__ ladp, int numTok)
{
    // B fragments (64x128 weights as bf16 hi/lo), pre-laid-out in WMMA lane order:
    // fragment fr = t*4+s covers N = 16t..16t+15, K = 32s..32s+31.
    __shared__ __align__(32) __bf16 sBhi[16 * 32 * 16];
    __shared__ __align__(32) __bf16 sBlo[16 * 32 * 16];
    __shared__ float sBias[NP];
    __shared__ float sLog[8 * 32 * LOG_STRIDE];   // per-wave 32 tokens x 64 logits

    const int tid = threadIdx.x;
    if (tid < NP) sBias[tid] = brow(tid, bw, bh, bd, bl);

    // Stage weight fragments: element i of lane L's v16bf for step s maps to
    // K = 32s + (i<8 ? 8*(L>>4)+i : 16 + 8*(L>>4) + (i-8)), N = 16t + (L&15).
    for (int e = tid; e < 16 * 32 * 16; e += 256) {
        int i  = e & 15;
        int L  = (e >> 4) & 31;
        int fr = e >> 9;              // fragment id 0..15
        int s  = fr & 3, t = fr >> 2;
        int h  = L >> 4;
        int n  = 16 * t + (L & 15);
        int k  = 32 * s + ((i < 8) ? (8 * h + i) : (16 + 8 * h + (i - 8)));
        float w = wrow(n, k, Ww, Wh, Wd, Wl);
        __bf16 hi = (__bf16)w;
        sBhi[e] = hi;
        sBlo[e] = (__bf16)(w - (float)hi);
    }
    __syncthreads();

    const int lane = tid & 31;
    const int wv   = tid >> 5;
    const int m    = lane & 15;
    const int half = lane >> 4;
    const int nTiles = (numTok + 31) >> 5;
    float* wLog = &sLog[wv * 32 * LOG_STRIDE];

    // Keep the hi-part of B resident in registers (16 fragments = 128 VGPRs);
    // the lo-part is streamed from LDS per use to stay under 256 VGPRs.
    v16bf Bhi[16];
    #pragma unroll
    for (int fr = 0; fr < 16; ++fr)
        Bhi[fr] = *(const v16bf*)&sBhi[(fr * 32 + lane) * 16];

    for (int tile = blockIdx.x * 8 + wv; tile < nTiles; tile += gridDim.x * 8) {
        const int base = tile * 32;

        // early-issue the per-token input load for the spline phase
        const int tokS = base + lane;
        const float x = (tokS < numTok) ? inp[tokS] : 0.f;

        // ---- GEMM: two 16-token x 64-param tiles via bf16 hi/lo split WMMA ----
        for (int sub = 0; sub < 2; ++sub) {
            int tok = base + sub * 16 + m;
            if (tok >= numTok) tok = numTok - 1;   // clamp (stores are guarded)
            const float* rowp = cond + (size_t)tok * DM;

            // issue all 16 A loads up front for memory-level parallelism
            float4 f[16];
            #pragma unroll
            for (int s = 0; s < 4; ++s) {
                int k0 = 32 * s + 8 * half;
                f[s * 4 + 0] = *(const float4*)(rowp + k0);
                f[s * 4 + 1] = *(const float4*)(rowp + k0 + 4);
                f[s * 4 + 2] = *(const float4*)(rowp + k0 + 16);
                f[s * 4 + 3] = *(const float4*)(rowp + k0 + 20);
            }

            v8f acc[4];
            #pragma unroll
            for (int t = 0; t < 4; ++t) acc[t] = (v8f){0.f,0.f,0.f,0.f,0.f,0.f,0.f,0.f};

            #pragma unroll
            for (int s = 0; s < 4; ++s) {
                // convert this K-step's A chunk; staging f32 regs die afterwards
                float v[16] = {f[s*4+0].x, f[s*4+0].y, f[s*4+0].z, f[s*4+0].w,
                               f[s*4+1].x, f[s*4+1].y, f[s*4+1].z, f[s*4+1].w,
                               f[s*4+2].x, f[s*4+2].y, f[s*4+2].z, f[s*4+2].w,
                               f[s*4+3].x, f[s*4+3].y, f[s*4+3].z, f[s*4+3].w};
                v16bf ahi, alo;
                #pragma unroll
                for (int i = 0; i < 16; ++i) {
                    __bf16 hi = (__bf16)v[i];
                    ahi[i] = hi;
                    alo[i] = (__bf16)(v[i] - (float)hi);
                }
                #pragma unroll
                for (int t = 0; t < 4; ++t) {
                    v16bf blo = *(const v16bf*)&sBlo[((t * 4 + s) * 32 + lane) * 16];
                    acc[t] = __builtin_amdgcn_wmma_f32_16x16x32_bf16(
                        false, ahi, false, Bhi[t * 4 + s], (short)0, acc[t], false, false);
                    acc[t] = __builtin_amdgcn_wmma_f32_16x16x32_bf16(
                        false, alo, false, Bhi[t * 4 + s], (short)0, acc[t], false, false);
                    acc[t] = __builtin_amdgcn_wmma_f32_16x16x32_bf16(
                        false, ahi, false, blo, (short)0, acc[t], false, false);
                }
            }

            // D layout: VGPR r -> (M = r + 8*half, N = 16t + (lane&15))
            #pragma unroll
            for (int t = 0; t < 4; ++t) {
                #pragma unroll
                for (int r = 0; r < 8; ++r) {
                    int mm = r + 8 * half;
                    wLog[(sub * 16 + mm) * LOG_STRIDE + (16 * t + m)] = acc[t][r];
                }
            }
        }
        // LDS from same wave is in-order; make the cross-lane handoff explicit.
        asm volatile("s_wait_dscnt 0" ::: "memory");

        // ---- spline: one token per lane ----
        if (tokS < numTok) {
            const float* zrow = &wLog[lane * LOG_STRIDE];
            bool inside = (x >= -TB) && (x <= TB);
            float xc = fminf(fmaxf(x, -TB), TB);

            // widths: double softmax, cumulative scan + bin search/gather
            float a[16];
            #pragma unroll
            for (int j = 0; j < 16; ++j) a[j] = zrow[j] + sBias[j];
            float m1 = a[0];
            #pragma unroll
            for (int j = 1; j < 16; ++j) m1 = fmaxf(m1, a[j]);
            float s1 = 0.f;
            #pragma unroll
            for (int j = 0; j < 16; ++j) { a[j] = expf(a[j] - m1); s1 += a[j]; }
            float is1 = 1.f / s1;          // max of first softmax == 1/s1
            float s2 = 0.f;
            #pragma unroll
            for (int j = 0; j < 16; ++j) { a[j] = expf((a[j] - 1.f) * is1); s2 += a[j]; }
            float cs = (1.f - 16.f * MBW) / s2;

            float C = 0.f; int idx = 0; float in_cw = -TB, in_bw = 1.f;
            #pragma unroll
            for (int j = 0; j < 16; ++j) {
                float wj = MBW + cs * a[j];
                float Cn = C + wj;
                float lo = (j == 0)  ? -TB : fmaf(2.f * TB, C,  -TB);
                float hi = (j == 15) ?  TB : fmaf(2.f * TB, Cn, -TB);
                float hiloc = (j == 15) ? (TB + SEPS) : hi;
                bool sel = (xc >= lo) && (xc < hiloc);
                if (sel) { idx = j; in_cw = lo; in_bw = hi - lo; }
                C = Cn;
            }

            // heights: same double softmax; gather at idx
            #pragma unroll
            for (int j = 0; j < 16; ++j) a[j] = zrow[16 + j] + sBias[16 + j];
            m1 = a[0];
            #pragma unroll
            for (int j = 1; j < 16; ++j) m1 = fmaxf(m1, a[j]);
            s1 = 0.f;
            #pragma unroll
            for (int j = 0; j < 16; ++j) { a[j] = expf(a[j] - m1); s1 += a[j]; }
            is1 = 1.f / s1;
            s2 = 0.f;
            #pragma unroll
            for (int j = 0; j < 16; ++j) { a[j] = expf((a[j] - 1.f) * is1); s2 += a[j]; }
            float csh = (1.f - 16.f * MBH) / s2;

            float Ch = 0.f; float in_ch = -TB, in_h = 1.f;
            #pragma unroll
            for (int j = 0; j < 16; ++j) {
                float hj = MBH + csh * a[j];
                float Cn = Ch + hj;
                float lo = (j == 0)  ? -TB : fmaf(2.f * TB, Ch, -TB);
                float hi = (j == 15) ?  TB : fmaf(2.f * TB, Cn, -TB);
                if (j == idx) { in_ch = lo; in_h = hi - lo; }
                Ch = Cn;
            }

            // derivatives: d[j] = 1 at the padded edges, else MDER + sp(sp(zd))
            float dk = 1.f, dk1 = 1.f;
            if (idx > 0)  dk  = MDER + splus(splus(zrow[31 + idx] + sBias[31 + idx]));
            if (idx < 15) dk1 = MDER + splus(splus(zrow[32 + idx] + sBias[32 + idx]));

            // lambda: 0.95*sigmoid(sigmoid(zl)) + 0.025
            float lam = 0.95f * sigm(sigm(zrow[47 + idx] + sBias[47 + idx])) + 0.025f;

            float in_delta = in_h / in_bw;
            float wb_ = sqrtf(dk / dk1);                       // wa == 1
            float wc_ = (lam * dk + (1.f - lam) * wb_ * dk1) / in_delta;
            float ya = in_ch, yb = in_h + in_ch;
            float yc = ((1.f - lam) * ya + lam * wb_ * yb) / ((1.f - lam) + lam * wb_);
            float theta = (xc - in_cw) / in_bw;
            bool mle = (theta <= lam);
            float num = mle ? (ya * (lam - theta) + wc_ * yc * theta)
                            : (wc_ * yc * (1.f - theta) + wb_ * yb * (theta - lam));
            float den = mle ? ((lam - theta) + wc_ * theta)
                            : (wc_ * (1.f - theta) + wb_ * (theta - lam));
            float sout = num / den;
            float dnum = (mle ? (wc_ * lam * (yc - ya))
                              : (wb_ * wc_ * (1.f - lam) * (yb - yc))) / in_bw;
            float lad = logf(dnum) - 2.f * logf(fabsf(den));

            outp[tokS] = inside ? sout : x;
            ladp[tokS] = inside ? lad : 0.f;
        }
    }
}

extern "C" void kernel_launch(void* const* d_in, const int* in_sizes, int n_in,
                              void* d_out, int out_size, void* d_ws, size_t ws_size,
                              hipStream_t stream) {
    const float* inp  = (const float*)d_in[0];
    const float* cond = (const float*)d_in[1];
    const float* Ww   = (const float*)d_in[2];
    const float* bw   = (const float*)d_in[3];
    const float* Wh   = (const float*)d_in[4];
    const float* bh   = (const float*)d_in[5];
    const float* Wd   = (const float*)d_in[6];
    const float* bd   = (const float*)d_in[7];
    const float* Wl   = (const float*)d_in[8];
    const float* bl   = (const float*)d_in[9];
    const int numTok = in_sizes[0];           // B*S
    float* outp = (float*)d_out;
    float* ladp = outp + numTok;

    int nTiles = (numTok + 31) / 32;          // 32 tokens per wave-iteration
    int blocks = (nTiles + 7) / 8;            // 8 waves per block
    if (blocks < 1) blocks = 1;
    rls_flow_kernel<<<blocks, 256, 0, stream>>>(inp, cond, Ww, bw, Wh, bh,
                                                Wd, bd, Wl, bl, outp, ladp, numTok);
}
